// ComplexMultiheadAttention_5248450035811
// MI455X (gfx1250) — matrix operand order repaired and checked by reading, not compiled, not run on hardware
//
#include <hip/hip_runtime.h>
#include <hip/hip_bf16.h>
#include <math.h>

// Problem dims (fixed by reference)
#define NB 8
#define NS 1024
#define NF 512
#define NE 512
#define NH 8
#define NC 64
#define NM (NB * NS)   // 8192 rows of (B*S)

typedef __attribute__((ext_vector_type(16))) _Float16 v16h;
typedef __attribute__((ext_vector_type(8)))  float    v8f;

static __device__ __forceinline__ v8f wmma_f16(v16h a, v16h b, v8f c) {
  // D = A(16x32 f16) * B(32x16 f16) + C(16x16 f32)
  return __builtin_amdgcn_wmma_f32_16x16x32_f16(
      /*neg_a=*/false, a, /*neg_b=*/false, b,
      /*c_mod=*/(short)0, c, /*reuse_a=*/false, /*reuse_b=*/false);
}

// ---- fragment loaders (wave32; this file's contiguous k order, same for A and B) ----
// A (16x32, f16): lane = hf*16+r holds row r; elem j -> K = hf*16 + j
static __device__ __forceinline__ v16h ldA_f32(const float* p, int ld, int row0, int k0) {
  int lane = threadIdx.x & 31, r = lane & 15, hf = lane >> 4;
  const float* q = p + (size_t)(row0 + r) * ld + k0;
  v16h a;
#pragma unroll
  for (int j = 0; j < 8; ++j) {
    a[j]     = (_Float16)q[hf * 16 + j];
    a[j + 8] = (_Float16)q[hf * 16 + 8 + j];
  }
  return a;
}
static __device__ __forceinline__ v16h ldA_f16(const _Float16* p, int ld, int row0, int k0) {
  int lane = threadIdx.x & 31, r = lane & 15, hf = lane >> 4;
  const _Float16* q = p + (size_t)(row0 + r) * ld + k0;
  v16h a;
#pragma unroll
  for (int j = 0; j < 8; ++j) {
    a[j]     = q[hf * 16 + j];
    a[j + 8] = q[hf * 16 + 8 + j];
  }
  return a;
}
// B (32x16, f16), element (k, n): lane holds col n = r, K = hf*16 + j.
// "Transposed" source: B[k][n] = src[n][k]  (src row-major n x k, contiguous in j)
static __device__ __forceinline__ v16h ldB_T_f32(const float* p, int ld, int n0, int k0) {
  int lane = threadIdx.x & 31, r = lane & 15, hf = lane >> 4;
  const float* q = p + (size_t)(n0 + r) * ld + k0 + hf * 16;
  v16h b;
#pragma unroll
  for (int j = 0; j < 16; ++j) b[j] = (_Float16)q[j];
  return b;
}
static __device__ __forceinline__ v16h ldB_T_f16(const _Float16* p, int ld, int n0, int k0) {
  int lane = threadIdx.x & 31, r = lane & 15, hf = lane >> 4;
  const _Float16* q = p + (size_t)(n0 + r) * ld + k0 + hf * 16;
  v16h b;
#pragma unroll
  for (int j = 0; j < 16; ++j) b[j] = q[j];
  return b;
}

// ---------------- Stage 1: complex QKV projection ----------------
// y = x @ w^T + b, complex; q,k stored f16 head-layout (B,H,S,C),
// v stored TRANSPOSED (B,H,C,S) so stage 3 B-fragments are contiguous.
__global__ void cmha_proj(const float* __restrict__ xr, const float* __restrict__ xi,
                          const float* wqr, const float* wqi, const float* bqr, const float* bqi,
                          const float* wkr, const float* wki, const float* bkr, const float* bki,
                          const float* wvr, const float* wvi, const float* bvr, const float* bvi,
                          _Float16* qrO, _Float16* qiO, _Float16* krO, _Float16* kiO,
                          _Float16* vrO, _Float16* viO) {
  int m0 = blockIdx.x * 16;
  int e0 = blockIdx.y * 16;
  int w  = blockIdx.z;  // 0=q, 1=k, 2=v
  const float* wr = (w == 0) ? wqr : (w == 1) ? wkr : wvr;
  const float* wi = (w == 0) ? wqi : (w == 1) ? wki : wvi;
  const float* br = (w == 0) ? bqr : (w == 1) ? bkr : bvr;
  const float* bi = (w == 0) ? bqi : (w == 1) ? bki : bvi;
  _Float16* oR = (w == 0) ? qrO : (w == 1) ? krO : vrO;
  _Float16* oI = (w == 0) ? qiO : (w == 1) ? kiO : viO;
  bool vT = (w == 2);

  v8f aRR = {}, aII = {}, aRI = {}, aIR = {};
  for (int kk = 0; kk < NF; kk += 32) {
    v16h axr = ldA_f32(xr, NF, m0, kk);
    v16h axi = ldA_f32(xi, NF, m0, kk);
    v16h bwr = ldB_T_f32(wr, NF, e0, kk);
    v16h bwi = ldB_T_f32(wi, NF, e0, kk);
    aRR = wmma_f16(axr, bwr, aRR);
    aII = wmma_f16(axi, bwi, aII);
    aRI = wmma_f16(axr, bwi, aRI);
    aIR = wmma_f16(axi, bwr, aIR);
  }
  int lane = threadIdx.x & 31, r = lane & 15, hf = lane >> 4;
  int e = e0 + r;
  int h = e / NC, c = e % NC;
  float br_v = br[e], bi_v = bi[e];
#pragma unroll
  for (int j = 0; j < 8; ++j) {
    int m = m0 + hf * 8 + j;
    int b = m / NS, s = m % NS;
    size_t idx = vT ? ((((size_t)b * NH + h) * NC + c) * NS + s)
                    : ((((size_t)b * NH + h) * NS + s) * NC + c);
    oR[idx] = (_Float16)(aRR[j] - aII[j] + br_v);
    oI[idx] = (_Float16)(aRI[j] + aIR[j] + bi_v);
  }
}

// ---------------- Stage 2: scores (Hermitian), abs, softmax ----------------
// one wave per (b,h, 16 rows of s): sweeps all t, softmax in LDS,
// writes complex attn to d_out (float2) and f16 attn copy to workspace
__global__ void cmha_attn(const _Float16* __restrict__ qrH, const _Float16* __restrict__ qiH,
                          const _Float16* __restrict__ krH, const _Float16* __restrict__ kiH,
                          _Float16* __restrict__ attnH, float2* __restrict__ attnOut) {
  __shared__ _Float16 simAbs[16][NS];  // 32 KB
  __shared__ float red[32];

  int s0 = blockIdx.x * 16;
  int bh = blockIdx.z * NH + blockIdx.y;
  const _Float16* qr = qrH + (size_t)bh * NS * NC;
  const _Float16* qi = qiH + (size_t)bh * NS * NC;
  const _Float16* kr = krH + (size_t)bh * NS * NC;
  const _Float16* ki = kiH + (size_t)bh * NS * NC;

  // q fragments for this row block, reused across all t tiles
  v16h aQr0 = ldA_f16(qr, NC, s0, 0),  aQr1 = ldA_f16(qr, NC, s0, 32);
  v16h aQi0 = ldA_f16(qi, NC, s0, 0),  aQi1 = ldA_f16(qi, NC, s0, 32);

  int lane = threadIdx.x & 31, r = lane & 15, hf = lane >> 4;
  const float invDiv = 0.0883883476483f;  // 1/sqrt(2*C) = 1/sqrt(128)

  for (int t0 = 0; t0 < NS; t0 += 16) {
    if (t0 + 16 < NS) {  // prefetch next k tile (global_prefetch_b8)
      __builtin_prefetch(kr + (size_t)(t0 + 16 + r) * NC, 0, 0);
      __builtin_prefetch(ki + (size_t)(t0 + 16 + r) * NC, 0, 0);
    }
    v8f accR = {}, accI1 = {}, accI2 = {};
    // K chunk 0..31 (c dim)
    v16h bKr = ldB_T_f16(kr, NC, t0, 0);
    v16h bKi = ldB_T_f16(ki, NC, t0, 0);
    accR  = wmma_f16(aQr0, bKr, accR);
    accR  = wmma_f16(aQi0, bKi, accR);   // + qi*ki   (Hermitian)
    accI1 = wmma_f16(aQi0, bKr, accI1);  // + qi*kr
    accI2 = wmma_f16(aQr0, bKi, accI2);  // (subtract at end)
    // K chunk 32..63
    bKr = ldB_T_f16(kr, NC, t0, 32);
    bKi = ldB_T_f16(ki, NC, t0, 32);
    accR  = wmma_f16(aQr1, bKr, accR);
    accR  = wmma_f16(aQi1, bKi, accR);
    accI1 = wmma_f16(aQi1, bKr, accI1);
    accI2 = wmma_f16(aQr1, bKi, accI2);
#pragma unroll
    for (int j = 0; j < 8; ++j) {
      int row = hf * 8 + j;
      float re = accR[j], im = accI1[j] - accI2[j];
      simAbs[row][t0 + r] = (_Float16)(sqrtf(re * re + im * im) * invDiv);
    }
  }
  __syncthreads();

  // softmax over t: lane owns row = lane&15, half = lane>>4 of the 1024 cols
  int row = r;
  int cbase = hf * (NS / 2);
  float pmax = -1e30f;
  for (int cc = 0; cc < NS / 2; ++cc)
    pmax = fmaxf(pmax, (float)simAbs[row][cbase + cc]);
  red[lane] = pmax;
  __syncthreads();
  float rmax = fmaxf(red[row], red[row + 16]);
  __syncthreads();
  float psum = 0.f;
  for (int cc = 0; cc < NS / 2; ++cc) {
    float p = __expf((float)simAbs[row][cbase + cc] - rmax);
    simAbs[row][cbase + cc] = (_Float16)p;
    psum += p;
  }
  red[lane] = psum;
  __syncthreads();
  float inv = 1.f / (red[row] + red[row + 16]);

  size_t rowbase = ((size_t)bh * NS + (s0 + row)) * NS;
  for (int cc = 0; cc < NS / 2; ++cc) {
    int t = cbase + cc;
    float p = (float)simAbs[row][t] * inv;
    attnH[rowbase + t] = (_Float16)p;
    attnOut[rowbase + t] = make_float2(p, 0.f);  // complex (p, 0), b64 store
  }
}

// ---------------- Stage 3: av = attn @ conj(v) ----------------
// v is stored transposed (B,H,C,S) -> B fragments are contiguous ldB_T loads
__global__ void cmha_av(const _Float16* __restrict__ attnH,
                        const _Float16* __restrict__ vrT, const _Float16* __restrict__ viT,
                        _Float16* __restrict__ cR, _Float16* __restrict__ cI) {
  int s0 = blockIdx.x * 16;
  int c0 = blockIdx.y * 16;
  int bh = blockIdx.z;
  const _Float16* A  = attnH + (size_t)bh * NS * NS;
  const _Float16* vr = vrT + (size_t)bh * NC * NS;
  const _Float16* vi = viT + (size_t)bh * NC * NS;

  v8f accR = {}, accI = {};
  for (int t0 = 0; t0 < NS; t0 += 32) {
    if (t0 + 32 < NS) {  // prefetch next attn A rows
      int lane = threadIdx.x & 31;
      __builtin_prefetch(A + (size_t)(s0 + (lane & 15)) * NS + t0 + 32, 0, 0);
    }
    v16h aA  = ldA_f16(A, NS, s0, t0);
    v16h bVr = ldB_T_f16(vr, NS, c0, t0);
    v16h bVi = ldB_T_f16(vi, NS, c0, t0);
    accR = wmma_f16(aA, bVr, accR);
    accI = wmma_f16(aA, bVi, accI);  // negate at store: conj(v)
  }
  int lane = threadIdx.x & 31, r = lane & 15, hf = lane >> 4;
  int b = bh / NH, h = bh % NH;
  int e = h * NC + c0 + r;
#pragma unroll
  for (int j = 0; j < 8; ++j) {
    int s = s0 + hf * 8 + j;
    size_t m = (size_t)b * NS + s;
    cR[m * NE + e] = (_Float16)accR[j];
    cI[m * NE + e] = (_Float16)(-accI[j]);
  }
}

// ---------------- Stage 4: out = concat @ wo^T + bo ----------------
__global__ void cmha_oproj(const _Float16* __restrict__ cR, const _Float16* __restrict__ cI,
                           const float* __restrict__ wor, const float* __restrict__ woi,
                           const float* __restrict__ bor, const float* __restrict__ boi,
                           float2* __restrict__ out) {
  int m0 = blockIdx.x * 16;
  int f0 = blockIdx.y * 16;
  v8f aRR = {}, aII = {}, aRI = {}, aIR = {};
  for (int e0 = 0; e0 < NE; e0 += 32) {
    v16h aCr = ldA_f16(cR, NE, m0, e0);
    v16h aCi = ldA_f16(cI, NE, m0, e0);
    v16h bWr = ldB_T_f32(wor, NE, f0, e0);
    v16h bWi = ldB_T_f32(woi, NE, f0, e0);
    aRR = wmma_f16(aCr, bWr, aRR);
    aII = wmma_f16(aCi, bWi, aII);
    aRI = wmma_f16(aCr, bWi, aRI);
    aIR = wmma_f16(aCi, bWr, aIR);
  }
  int lane = threadIdx.x & 31, r = lane & 15, hf = lane >> 4;
  int f = f0 + r;
  float br = bor[f], bi = boi[f];
#pragma unroll
  for (int j = 0; j < 8; ++j) {
    int m = m0 + hf * 8 + j;
    out[(size_t)m * NF + f] = make_float2(aRR[j] - aII[j] + br, aRI[j] + aIR[j] + bi);
  }
}

extern "C" void kernel_launch(void* const* d_in, const int* in_sizes, int n_in,
                              void* d_out, int out_size, void* d_ws, size_t ws_size,
                              hipStream_t stream) {
  (void)in_sizes; (void)n_in; (void)out_size; (void)ws_size;
  const float* x_r  = (const float*)d_in[0];
  const float* x_i  = (const float*)d_in[1];
  const float* wq_r = (const float*)d_in[2];
  const float* wq_i = (const float*)d_in[3];
  const float* bq_r = (const float*)d_in[4];
  const float* bq_i = (const float*)d_in[5];
  const float* wk_r = (const float*)d_in[6];
  const float* wk_i = (const float*)d_in[7];
  const float* bk_r = (const float*)d_in[8];
  const float* bk_i = (const float*)d_in[9];
  const float* wv_r = (const float*)d_in[10];
  const float* wv_i = (const float*)d_in[11];
  const float* bv_r = (const float*)d_in[12];
  const float* bv_i = (const float*)d_in[13];
  const float* wo_r = (const float*)d_in[14];
  const float* wo_i = (const float*)d_in[15];
  const float* bo_r = (const float*)d_in[16];
  const float* bo_i = (const float*)d_in[17];

  float2* out = (float2*)d_out;                      // (B,S,F) complex
  float2* attnOut = out + (size_t)NM * NF;           // (B,H,S,S) complex

  // workspace layout (all f16)
  _Float16* ws = (_Float16*)d_ws;
  const size_t NHSC = (size_t)NB * NH * NS * NC;     // 4,194,304
  const size_t NA   = (size_t)NB * NH * NS * NS;     // 67,108,864
  const size_t NME  = (size_t)NM * NE;               // 4,194,304
  _Float16* qr = ws;
  _Float16* qi = qr + NHSC;
  _Float16* kr = qi + NHSC;
  _Float16* ki = kr + NHSC;
  _Float16* vr = ki + NHSC;   // transposed (B,H,C,S)
  _Float16* vi = vr + NHSC;   // transposed (B,H,C,S)
  _Float16* attnH = vi + NHSC;
  _Float16* cR = attnH + NA;
  _Float16* cI = cR + NME;

  cmha_proj<<<dim3(NM / 16, NE / 16, 3), 32, 0, stream>>>(
      x_r, x_i, wq_r, wq_i, bq_r, bq_i, wk_r, wk_i, bk_r, bk_i,
      wv_r, wv_i, bv_r, bv_i, qr, qi, kr, ki, vr, vi);

  cmha_attn<<<dim3(NS / 16, NH, NB), 32, 0, stream>>>(qr, qi, kr, ki, attnH, attnOut);

  cmha_av<<<dim3(NS / 16, NC / 16, NB * NH), 32, 0, stream>>>(attnH, vr, vi, cR, cI);

  cmha_oproj<<<dim3(NM / 16, NF / 16), 32, 0, stream>>>(cR, cI, wo_r, wo_i, bo_r, bo_i, out);
}
